// STFTfourierKanMLPLiteDGCNN_42709154791523
// MI455X (gfx1250) — matrix operand order repaired and checked by reading, not compiled, not run on hardware
//
#include <hip/hip_runtime.h>
#include <hip/hip_bf16.h>

// ---------------------------------------------------------------------------
// Problem constants (from reference)
// ---------------------------------------------------------------------------
#define BATCH   32
#define NPTS    1024
#define KNN_K   20
#define EMB     1024
#define OUTC    7
// KAN1: window 52, stride 20, grid 7 over 128 dims -> 4 windows
#define WS1     52
#define ST1     20
#define G1      7
#define NW1     4
#define FEAT1   (NW1 * WS1 * G1)        // 1456
#define KDIM1   (2 * FEAT1)             // 2912 = 91 * 32
#define KCH1    (KDIM1 / 32)            // 91
// KAN2: window 197, stride 7, grid 6 over 2048 dims -> 265 windows
#define WS2     197
#define ST2     7
#define G2      6
#define NW2     265

#define NPOINTS (BATCH * NPTS)          // 32768
#define NEDGES  (NPOINTS * KNN_K)       // 655360

// GEMM macro-tiling: each wave computes (MT*16) x (NT*16) of C
#define MT 4
#define NT 4
#define NMAC (EMB / (16 * NT))          // 16 macro column tiles
#define MMAC (NPOINTS / (16 * MT))      // 512 macro row tiles
#define TSTRIDE (KCH1 * 512)            // halves between consecutive 16-row tiles

typedef __attribute__((ext_vector_type(16))) _Float16 v16h;
typedef __attribute__((ext_vector_type(8)))  _Float16 v8h;
typedef __attribute__((ext_vector_type(8)))  float    v8f;

// ---------------------------------------------------------------------------
// 1) brute-force kNN per graph (exclude self). block=(128), grid=(32, 8)
// ---------------------------------------------------------------------------
__global__ __launch_bounds__(128)
void knn_kernel(const float* __restrict__ pos, int* __restrict__ knn) {
  __shared__ float sx[NPTS], sy[NPTS], sz[NPTS];
  const int b = blockIdx.x;
  const int q = blockIdx.y * 128 + threadIdx.x;
  const float* P = pos + (size_t)b * NPTS * 3;
  for (int j = threadIdx.x; j < NPTS; j += 128) {
    sx[j] = P[j * 3 + 0];
    sy[j] = P[j * 3 + 1];
    sz[j] = P[j * 3 + 2];
  }
  __syncthreads();
  const float qx = sx[q], qy = sy[q], qz = sz[q];
  float bd[KNN_K];
  int   bi[KNN_K];
  for (int i = 0; i < KNN_K; ++i) { bd[i] = 3.0e38f; bi[i] = 0; }
  for (int j = 0; j < NPTS; ++j) {
    if (j == q) continue;
    const float dx = sx[j] - qx, dy = sy[j] - qy, dz = sz[j] - qz;
    const float d = dx * dx + dy * dy + dz * dz;
    if (d < bd[KNN_K - 1]) {
      int p = KNN_K - 1;
      while (p > 0 && bd[p - 1] > d) { bd[p] = bd[p - 1]; bi[p] = bi[p - 1]; --p; }
      bd[p] = d; bi[p] = j;
    }
  }
  int* out = knn + ((size_t)b * NPTS + q) * KNN_K;
  for (int i = 0; i < KNN_K; ++i) out[i] = bi[i];
}

// ---------------------------------------------------------------------------
// 2) BatchNorm batch statistics of h = e @ W1 + b1 over all edges.
// ---------------------------------------------------------------------------
__global__ __launch_bounds__(64)
void edge_stats_kernel(const float* __restrict__ pos, const int* __restrict__ knn,
                       const float* __restrict__ W1, const float* __restrict__ b1,
                       float* __restrict__ stats) {
  const int c = threadIdx.x;
  float w[6];
#pragma unroll
  for (int i = 0; i < 6; ++i) w[i] = W1[i * 64 + c];
  const float bb = b1[c];
  float s = 0.0f, s2 = 0.0f;
  const int e0 = blockIdx.x * 640;
  for (int e = e0; e < e0 + 640; ++e) {
    const int pt = e / KNN_K;
    const int bg = pt >> 10;
    const int j  = knn[e];
    const float* Pi = pos + (size_t)pt * 3;
    const float* Pj = pos + ((size_t)((bg << 10) + j)) * 3;
    const float xi = Pi[0], yi = Pi[1], zi = Pi[2];
    const float h = xi * w[0] + yi * w[1] + zi * w[2] +
                    (Pj[0] - xi) * w[3] + (Pj[1] - yi) * w[4] + (Pj[2] - zi) * w[5] + bb;
    s += h; s2 += h * h;
  }
  atomicAdd(&stats[c], s);
  atomicAdd(&stats[64 + c], s2);
}

__global__ __launch_bounds__(64)
void finalize_norm_kernel(const float* __restrict__ stats, const float* __restrict__ gamma,
                          const float* __restrict__ beta, float* __restrict__ norm) {
  const int c = threadIdx.x;
  const float inv = 1.0f / (float)NEDGES;
  const float mu  = stats[c] * inv;
  const float var = stats[64 + c] * inv - mu * mu;
  const float sc  = gamma[c] * rsqrtf(var + 1e-5f);
  norm[c]      = sc;
  norm[64 + c] = beta[c] - mu * sc;
}

// ---------------------------------------------------------------------------
// 3) Edge MLP (Linear->BN->ReLU->Linear) + max aggregation over K.
// ---------------------------------------------------------------------------
__global__ __launch_bounds__(128)
void edge_mlp_max_kernel(const float* __restrict__ pos, const int* __restrict__ knn,
                         const float* __restrict__ W1, const float* __restrict__ b1,
                         const float* __restrict__ norm, const float* __restrict__ W2,
                         const float* __restrict__ b2, float* __restrict__ x1) {
  __shared__ float hs[64];
  __shared__ float e6[6];
  const int pt  = blockIdx.x;
  const int tid = threadIdx.x;
  const int bg  = pt >> 10;
  const float* Pi = pos + (size_t)pt * 3;
  const float xi = Pi[0], yi = Pi[1], zi = Pi[2];
  const float bias2 = b2[tid];
  float acc = -3.0e38f;
  for (int k = 0; k < KNN_K; ++k) {
    __syncthreads();
    if (tid == 0) {
      const int j = knn[(size_t)pt * KNN_K + k];
      const float* Pj = pos + ((size_t)((bg << 10) + j)) * 3;
      e6[0] = xi; e6[1] = yi; e6[2] = zi;
      e6[3] = Pj[0] - xi; e6[4] = Pj[1] - yi; e6[5] = Pj[2] - zi;
    }
    __syncthreads();
    if (tid < 64) {
      float h = b1[tid];
#pragma unroll
      for (int i = 0; i < 6; ++i) h += e6[i] * W1[i * 64 + tid];
      h = h * norm[tid] + norm[64 + tid];
      hs[tid] = fmaxf(h, 0.0f);
    }
    __syncthreads();
    float o = bias2;
    for (int j = 0; j < 64; ++j) o += hs[j] * W2[j * 128 + tid];
    acc = fmaxf(acc, o);
  }
  x1[(size_t)pt * 128 + tid] = acc;
}

// ---------------------------------------------------------------------------
// 4) Build f16 Fourier feature matrix A [32768 x 2912] directly in the WMMA
//    A-fragment layout (16-bit A 16x32, ISA 7.12.2):
//      lane 0..15 : a[0..7]=K 0..7,  a[8..15]=K 16..23   (row M = lane)
//      lane 16..31: a[0..7]=K 8..15, a[8..15]=K 24..31   (row M = lane-16)
//    Packed: Apk[((mtile*91 + kchunk)*32 + lane)*16 + half]
//    Any 8-aligned run of 8 consecutive K columns maps to 8 consecutive
//    halves of ONE lane -> each thread emits two 16-byte stores.
// ---------------------------------------------------------------------------
__device__ __forceinline__ size_t a_frag_group8(int m, int col0 /* multiple of 8 */) {
  const int mtile = m >> 4, mr = m & 15;
  const int kchunk = col0 >> 5, kc = col0 & 31;   // kc in {0,8,16,24}
  int lane, half;
  if (kc == 0)       { lane = mr;      half = 0; }
  else if (kc == 8)  { lane = mr + 16; half = 0; }
  else if (kc == 16) { lane = mr;      half = 8; }
  else               { lane = mr + 16; half = 8; }
  return (((size_t)mtile * KCH1 + kchunk) << 9) + (size_t)lane * 16 + half;
}

__global__ __launch_bounds__(256)
void build_features_kernel(const float* __restrict__ x1, _Float16* __restrict__ Apk) {
  const int t = blockIdx.x * 256 + threadIdx.x;   // t < NPOINTS * (FEAT1/8)
  if (t >= NPOINTS * (FEAT1 / 8)) return;
  const int fg = t % (FEAT1 / 8);                 // group of 8 columns
  const int m  = t / (FEAT1 / 8);
  v8h cvals, svals;
#pragma unroll
  for (int u = 0; u < 8; ++u) {
    const int f = fg * 8 + u;
    const int g = f % G1;
    const int w = (f / G1) % WS1;
    const int n = f / (G1 * WS1);
    const float win = 1.0f - fabsf(2.0f * (float)w / (float)WS1 - 1.0f);  // periodic bartlett
    const float xv  = x1[(size_t)m * 128 + n * ST1 + w] * win;
    float sv, cv;
    __sincosf(xv * (float)(g + 1), &sv, &cv);
    cvals[u] = (_Float16)cv;
    svals[u] = (_Float16)sv;
  }
  *(v8h*)(Apk + a_frag_group8(m, fg * 8))         = cvals;   // cos block
  *(v8h*)(Apk + a_frag_group8(m, FEAT1 + fg * 8)) = svals;   // sin block
}

// ---------------------------------------------------------------------------
// 5) Pack c1 [2, 1024, 4, 52, 7] -> f16 B-fragment layout (16-bit B 32x16):
//      lane 0..15 : b[0..7]=K 0..7,  b[8..15]=K 16..23 of column N = lane
//      lane 16..31: b[0..7]=K 8..15, b[8..15]=K 24..31 of column N = lane-16
// ---------------------------------------------------------------------------
__global__ __launch_bounds__(256)
void pack_c1_kernel(const float* __restrict__ c1, _Float16* __restrict__ Bpk) {
  const int t = blockIdx.x * 256 + threadIdx.x;
  const int TOT = 2 * EMB * FEAT1;
  if (t >= TOT) return;
  int r = t;
  const int g = r % G1;  r /= G1;
  const int w = r % WS1; r /= WS1;
  const int n = r % NW1; r /= NW1;
  const int o = r % EMB;
  const int s = r / EMB;
  const int f = s * FEAT1 + (n * WS1 + w) * G1 + g;   // K row in [0, 2912)
  const int ntile = o >> 4, nr = o & 15;
  const int kchunk = f >> 5, kc = f & 31;
  const int lane = ((kc >> 3) & 1) ? nr + 16 : nr;
  const int half = (kc & 7) | ((kc >> 4) << 3);
  Bpk[(((size_t)ntile * KCH1 + kchunk) << 9) + (size_t)lane * 16 + half] = (_Float16)c1[t];
}

// ---------------------------------------------------------------------------
// 6) KAN1 GEMM: C[32768,1024] = A[32768,2912] x B[2912,1024] + bias.
//    Register-blocked 64x64 macro-tile per wave (4x4 WMMA tiles, 16
//    independent v_wmma_f32_16x16x32_f16 per k-chunk, 32 FLOP/byte from L2).
//    Tiles addressed with compile-time IOFFSETs off two base pointers so the
//    compiler emits immediate-offset global_load_b128 clauses (not FLAT).
//    A (191 MB) streams from HBM once -> prefetch 8 k-chunks (8 KB) ahead.
//    block = 256 (8 waves), grid = (512*16)/8 = 1024 blocks.
// ---------------------------------------------------------------------------
__global__ __launch_bounds__(256)
void kan1_wmma_gemm(const _Float16* __restrict__ Apk, const _Float16* __restrict__ Bpk,
                    const float* __restrict__ bias, float* __restrict__ C) {
  const int wave = (blockIdx.x * blockDim.x + threadIdx.x) >> 5;
  const int lane = threadIdx.x & 31;
  const int nmac = wave & (NMAC - 1);   // fast index -> A macro-tile L2 reuse
  const int mmac = wave / NMAC;

  const _Float16* __restrict__ Abase =
      Apk + (((size_t)(mmac * MT) * KCH1) << 9) + (size_t)lane * 16;
  const _Float16* __restrict__ Bbase =
      Bpk + (((size_t)(nmac * NT) * KCH1) << 9) + (size_t)lane * 16;

  v8f acc[MT][NT] = {};

  for (int kc = 0; kc < KCH1; ++kc) {
    v16h a[MT], b[NT];
#pragma unroll
    for (int i = 0; i < MT; ++i)
      a[i] = *(const v16h*)(Abase + (size_t)i * TSTRIDE);      // ioffset = i*93184 B
#pragma unroll
    for (int j = 0; j < NT; ++j)
      b[j] = *(const v16h*)(Bbase + (size_t)j * TSTRIDE);
#pragma unroll
    for (int i = 0; i < MT; ++i)                               // A streams from HBM:
      __builtin_prefetch(Abase + (size_t)i * TSTRIDE + 8 * 512, 0, 1);
#pragma unroll
    for (int i = 0; i < MT; ++i)
#pragma unroll
      for (int j = 0; j < NT; ++j)
        acc[i][j] = __builtin_amdgcn_wmma_f32_16x16x32_f16(false, a[i], false, b[j],
                                                           (short)0, acc[i][j],
                                                           false, false);
    Abase += 512;   // next 16x32 k-chunk (1 KB per tile)
    Bbase += 512;
  }

  // C/D layout: VGPR r -> M=r (lanes 0..15) / M=r+8 (lanes 16..31), N=lane&15.
  const int half = lane >> 4;
  const int cn   = lane & 15;
#pragma unroll
  for (int i = 0; i < MT; ++i) {
    const int row0 = ((mmac * MT + i) << 4) + half * 8;
#pragma unroll
    for (int j = 0; j < NT; ++j) {
      const int col = ((nmac * NT + j) << 4) + cn;
      const float bv = bias[col];
#pragma unroll
      for (int r = 0; r < 8; ++r)
        C[(size_t)(row0 + r) * EMB + col] = acc[i][j][r] + bv;
    }
  }
}

// ---------------------------------------------------------------------------
// 7) Global max + mean pool over N=1024 points. grid=(4,32), block=256.
// ---------------------------------------------------------------------------
__global__ __launch_bounds__(256)
void global_pool_kernel(const float* __restrict__ X, float* __restrict__ xcat) {
  const int c = blockIdx.x * 256 + threadIdx.x;   // channel 0..1023
  const int b = blockIdx.y;
  if (c >= EMB) return;
  const float* base = X + ((size_t)b * NPTS) * EMB + c;
  float mx = -3.0e38f, sm = 0.0f;
  for (int n = 0; n < NPTS; ++n) {
    const float v = base[(size_t)n * EMB];
    mx = fmaxf(mx, v);
    sm += v;
  }
  xcat[(size_t)b * (2 * EMB) + c]       = mx;
  xcat[(size_t)b * (2 * EMB) + EMB + c] = sm * (1.0f / (float)NPTS);
}

// ---------------------------------------------------------------------------
// 8) KAN2: [32, 2048] -> [32, 7]. block per graph, shared-atomic reduction.
// ---------------------------------------------------------------------------
__global__ __launch_bounds__(256)
void kan2_kernel(const float* __restrict__ xcat, const float* __restrict__ c2,
                 const float* __restrict__ bi2, float* __restrict__ out) {
  __shared__ float accs[OUTC];
  const int b = blockIdx.x;
  const int tid = threadIdx.x;
  if (tid < OUTC) accs[tid] = 0.0f;
  __syncthreads();
  float acc[OUTC];
#pragma unroll
  for (int o = 0; o < OUTC; ++o) acc[o] = 0.0f;
  const float* xrow = xcat + (size_t)b * (2 * EMB);
  const size_t stride_o = (size_t)NW2 * WS2 * G2;   // 313230
  const int NP = NW2 * WS2;                          // 52205 (n, w) pairs
  for (int p = tid; p < NP; p += 256) {
    const int n = p / WS2, w = p % WS2;
    const float win = 0.5f * (1.0f - __cosf(6.28318530717958647692f * (float)w / (float)WS2));
    const float xv  = xrow[n * ST2 + w] * win;
    const size_t base = (size_t)p * G2;
#pragma unroll
    for (int g = 0; g < G2; ++g) {
      const float ang = xv * (float)(g + 1);
      float sv, cv;
      __sincosf(ang, &sv, &cv);
#pragma unroll
      for (int o = 0; o < OUTC; ++o) {
        acc[o] += cv * c2[(size_t)o * stride_o + base + g] +
                  sv * c2[(size_t)(OUTC + o) * stride_o + base + g];
      }
    }
  }
#pragma unroll
  for (int o = 0; o < OUTC; ++o) atomicAdd(&accs[o], acc[o]);
  __syncthreads();
  if (tid < OUTC) out[b * OUTC + tid] = accs[tid] + bi2[tid];
}

// ---------------------------------------------------------------------------
// Launch
// ---------------------------------------------------------------------------
extern "C" void kernel_launch(void* const* d_in, const int* in_sizes, int n_in,
                              void* d_out, int out_size, void* d_ws, size_t ws_size,
                              hipStream_t stream) {
  (void)in_sizes; (void)n_in; (void)out_size; (void)ws_size;
  const float* pos    = (const float*)d_in[0];
  // d_in[1] = batch (implicit: equal-size sorted graphs)
  const float* W1     = (const float*)d_in[2];
  const float* b1     = (const float*)d_in[3];
  const float* gamma1 = (const float*)d_in[4];
  const float* beta1  = (const float*)d_in[5];
  const float* W2     = (const float*)d_in[6];
  const float* b2     = (const float*)d_in[7];
  const float* c1     = (const float*)d_in[8];
  const float* bi1    = (const float*)d_in[9];
  const float* c2     = (const float*)d_in[10];
  const float* bi2    = (const float*)d_in[11];
  float* out = (float*)d_out;

  char* ws = (char*)d_ws;
  size_t off = 0;
  auto take = [&](size_t bytes) -> char* {
    char* p = ws + off;
    off = (off + bytes + 255) & ~(size_t)255;
    return p;
  };
  float*    stats = (float*)take(512);                                   // 64 sum + 64 sumsq
  float*    norm  = (float*)take(512);                                   // 64 scale + 64 shift
  int*      knn   = (int*)take((size_t)NEDGES * sizeof(int));            // 2.6 MB
  float*    x1    = (float*)take((size_t)NPOINTS * 128 * sizeof(float)); // 16.8 MB
  _Float16* Apk   = (_Float16*)take((size_t)NPOINTS * KDIM1 * 2);        // 191 MB
  _Float16* Bpk   = (_Float16*)take((size_t)KDIM1 * EMB * 2);            // 6 MB
  float*    Xemb  = (float*)take((size_t)NPOINTS * EMB * sizeof(float)); // 134 MB
  float*    xcat  = (float*)take((size_t)BATCH * 2 * EMB * sizeof(float));

  hipMemsetAsync(stats, 0, 512, stream);

  knn_kernel<<<dim3(BATCH, NPTS / 128), 128, 0, stream>>>(pos, knn);
  edge_stats_kernel<<<NEDGES / 640, 64, 0, stream>>>(pos, knn, W1, b1, stats);
  finalize_norm_kernel<<<1, 64, 0, stream>>>(stats, gamma1, beta1, norm);
  edge_mlp_max_kernel<<<NPOINTS, 128, 0, stream>>>(pos, knn, W1, b1, norm, W2, b2, x1);

  pack_c1_kernel<<<(2 * EMB * FEAT1 + 255) / 256, 256, 0, stream>>>(c1, Bpk);
  build_features_kernel<<<NPOINTS * (FEAT1 / 8) / 256, 256, 0, stream>>>(x1, Apk);

  // (512 m-macros x 16 n-macros) waves / 8 waves per block = 1024 blocks
  kan1_wmma_gemm<<<(MMAC * NMAC) / 8, 256, 0, stream>>>(Apk, Bpk, bi1, Xemb);

  global_pool_kernel<<<dim3(EMB / 256, BATCH), 256, 0, stream>>>(Xemb, xcat);
  kan2_kernel<<<BATCH, 256, 0, stream>>>(xcat, c2, bi2, out);
}
